// RGBBlock_44779329028566
// MI455X (gfx1250) — matrix operand order repaired and checked by reading, not compiled, not run on hardware
//
#include <hip/hip_runtime.h>

typedef __bf16 bf16;
typedef __bf16 v16bf __attribute__((ext_vector_type(16)));
typedef float  v8f   __attribute__((ext_vector_type(8)));
typedef int    v4i   __attribute__((ext_vector_type(4)));

#define B_SZ    8
#define C_INCH  256
#define FILT    512
#define LAT     512
#define T_IN    8192
#define T_OUT   16384

#define MT      128   // o-rows per block
#define NT      32    // t-cols per block
#define KS      32    // K per WMMA step
#define NTB     256   // T_IN / NT

#define XPITCH  40    // LDS row pitch in bf16 (multiple of 8 -> 16B aligned rows)
#define MPITCH  132   // mid tile row pitch in floats

// ---------------------------------------------------------------------------
// gfx1250 async global->LDS copy (ASYNCcnt) with compile-safe fallback.
// Probe (round 2 diagnostic) shows param types: (int4 AS1*, int4 AS3*, Ii, Ii).
// ---------------------------------------------------------------------------
#if defined(__has_builtin)
#if __has_builtin(__builtin_amdgcn_global_load_async_to_lds_b128)
#define HAVE_ASYNC_LDS 1
#endif
#endif

#ifdef HAVE_ASYNC_LDS
#define ASYNC_CP16(g, l)                                                      \
    __builtin_amdgcn_global_load_async_to_lds_b128(                           \
        (__attribute__((address_space(1))) v4i*)(g),                          \
        (__attribute__((address_space(3))) v4i*)(l), 0, 0)
#if __has_builtin(__builtin_amdgcn_s_wait_asynccnt)
#define ASYNC_WAIT(n) __builtin_amdgcn_s_wait_asynccnt(n)
#else
#define ASYNC_WAIT(n) asm volatile("s_wait_asynccnt %0" ::"i"(n) : "memory")
#endif
#else
#define ASYNC_CP16(g, l) (*(uint4*)(l) = *(const uint4*)(g))
#define ASYNC_WAIT(n) ((void)0)
#endif

// ---------------------------------------------------------------------------
// A-fragment K index for V_WMMA_F32_16X16X32_BF16 (ISA 7.12.2, 16-bit A 16x32):
// lanes 0-15: VGPR0..3 -> K=0..7 (pairs), VGPR4..7 -> K=16..23; lanes 16-31: +8
// ---------------------------------------------------------------------------
__device__ __host__ inline int a_frag_k(int lane, int e) {
    int v = e >> 1;
    int base = (v < 4 ? 2 * v : 16 + 2 * (v - 4)) + ((lane & 16) ? 8 : 0);
    return base + (e & 1);
}

// ---------------------------------------------------------------------------
// style scales: s[b,f] = istyle[b,:] . W_style[f,:] + b_style[f] + 1
// ---------------------------------------------------------------------------
__global__ void k_style(const float* __restrict__ istyle,
                        const float* __restrict__ Wsty,
                        const float* __restrict__ bsty,
                        float* __restrict__ s_out) {
    int idx = blockIdx.x * blockDim.x + threadIdx.x;
    if (idx >= B_SZ * FILT) return;
    int b = idx >> 9, f = idx & 511;
    const float* iv = istyle + (size_t)b * LAT;
    const float* wv = Wsty + (size_t)f * LAT;
    float acc = 0.f;
    for (int l = 0; l < LAT; ++l) acc += iv[l] * wv[l];
    s_out[idx] = acc + bsty[f] + 1.0f;
}

// ---------------------------------------------------------------------------
// Per-batch modulated A-fragments: wmodFb[b][ot(32)][ks(16)][lane(32)][e(16)]
// value = W_mod[o,c] * s[b,c]   (modulation folded into the small operand)
// ---------------------------------------------------------------------------
__global__ void k_pack_wmod(const float* __restrict__ Wmod,
                            const float* __restrict__ s,
                            bf16* __restrict__ out) {
    int idx = blockIdx.x * blockDim.x + threadIdx.x;
    if (idx >= B_SZ * 32 * 16 * 32 * 16) return;
    int e    = idx & 15;
    int lane = (idx >> 4) & 31;
    int ks   = (idx >> 9) & 15;
    int ot   = (idx >> 13) & 31;
    int b    = idx >> 18;
    int m = lane & 15;
    int k = ks * KS + a_frag_k(lane, e);
    out[idx] = (bf16)(Wmod[(size_t)(ot * 16 + m) * FILT + k] * s[b * FILT + k]);
}

// ---------------------------------------------------------------------------
// Pack W_conv1 (512x256x3) into wconvF[tap(3)][ot(32)][ks(8)][lane(32)][e(16)]
// ---------------------------------------------------------------------------
__global__ void k_pack_wconv(const float* __restrict__ Wc, bf16* __restrict__ out) {
    int idx = blockIdx.x * blockDim.x + threadIdx.x;
    if (idx >= 3 * 32 * 8 * 32 * 16) return;
    int e    = idx & 15;
    int lane = (idx >> 4) & 31;
    int ks   = (idx >> 9) & 7;
    int ot   = (idx >> 12) & 31;
    int tap  = idx >> 17;
    int m = lane & 15;
    int c = ks * KS + a_frag_k(lane, e);
    out[idx] = (bf16)Wc[((size_t)(ot * 16 + m) * C_INCH + c) * 3 + tap];
}

// ---------------------------------------------------------------------------
// Transpose x (B,512,T) f32 -> xT (B,T,512) bf16 (c-contiguous rows)
// ---------------------------------------------------------------------------
__global__ __launch_bounds__(256) void k_tr_x(const float* __restrict__ x,
                                              bf16* __restrict__ xT) {
    __shared__ bf16 tile[32][33];
    int tb = blockIdx.x, cb = blockIdx.y, b = blockIdx.z;
    int l = threadIdx.x & 31, q = threadIdx.x >> 5;
    #pragma unroll
    for (int i = 0; i < 4; ++i) {
        int c = q + i * 8;
        tile[c][l] = (bf16)x[(size_t)(b * FILT + cb * 32 + c) * T_IN + tb * 32 + l];
    }
    __syncthreads();
    #pragma unroll
    for (int i = 0; i < 4; ++i) {
        int t = q + i * 8;
        xT[(size_t)(b * T_IN + tb * 32 + t) * FILT + cb * 32 + l] = tile[l][t];
    }
}

// ---------------------------------------------------------------------------
// Transpose prev_rgb (B,256,T) f32 -> prevT (B,T+2,256) bf16, halo rows at
// t=0 and t=T+1 zeroed separately (conv pad -> no bounds checks in main loop).
// ---------------------------------------------------------------------------
__global__ __launch_bounds__(256) void k_tr_prev(const float* __restrict__ p,
                                                 bf16* __restrict__ pT) {
    __shared__ bf16 tile[32][33];
    int tb = blockIdx.x, cb = blockIdx.y, b = blockIdx.z;
    int l = threadIdx.x & 31, q = threadIdx.x >> 5;
    #pragma unroll
    for (int i = 0; i < 4; ++i) {
        int c = q + i * 8;
        tile[c][l] = (bf16)p[(size_t)(b * C_INCH + cb * 32 + c) * T_IN + tb * 32 + l];
    }
    __syncthreads();
    #pragma unroll
    for (int i = 0; i < 4; ++i) {
        int t = q + i * 8;
        pT[(size_t)(b * (T_IN + 2) + tb * 32 + t + 1) * C_INCH + cb * 32 + l] = tile[l][t];
    }
}

__global__ void k_zero_pad(bf16* __restrict__ pT) {
    int idx = blockIdx.x * blockDim.x + threadIdx.x;
    if (idx >= B_SZ * 2 * C_INCH) return;
    int c = idx & 255;
    int r = (idx >> 8) & 1;
    int b = idx >> 9;
    pT[(size_t)(b * (T_IN + 2) + (r ? T_IN + 1 : 0)) * C_INCH + c] = (bf16)0.0f;
}

// ---------------------------------------------------------------------------
// Main fused kernel: (modulated-weight GEMM + conv-as-GEMM) + bias + 2x
// upsample. Block = 8 waves; tile = 128(o) x 32(t); grid = (256, 4, 8).
// Operand tiles streamed by double-buffered async global->LDS copies.
// ---------------------------------------------------------------------------
__global__ __launch_bounds__(256) void k_main(
    const bf16* __restrict__ xT, const bf16* __restrict__ prevT,
    const bf16* __restrict__ wmodFb, const bf16* __restrict__ wconvF,
    const float* __restrict__ bias, float* __restrict__ out,
    float* __restrict__ edgeF, float* __restrict__ edgeL) {

    __shared__ __align__(16) bf16  xs_s[2][NT * XPITCH];
    __shared__ __align__(16) bf16  pv_s[2][(NT + 2) * XPITCH];
    __shared__ __align__(16) float mid_s[NT * MPITCH];

    const int tid  = threadIdx.x;
    const int lane = tid & 31, w = tid >> 5;
    const int t_sub = w & 1, o_sub = w >> 1;       // waves: 4 (o) x 2 (t)
    const int tn = t_sub << 4;
    const int tb = blockIdx.x, ob = blockIdx.y, b = blockIdx.z;
    const int t0 = tb * NT;

    const int l16 = lane & 15, lh = lane >> 4;
    const int brow = tn + l16;          // B fragment: n = t column
    const int bcol = lh << 3;           // lanes 0-15 -> K 0-7,16-23, lanes 16-31 -> K 8-15,24-31
    const int ot0  = ob * 8 + o_sub * 2;

    v8f acc0 = {}; v8f acc1 = {};

    // async tile issue: one b128 (16B = 8 bf16) per thread, 4 segs per row
    auto issue_x = [&](int ks, int buf) {
        if (tid < NT * 4) {
            int r = tid >> 2, seg = tid & 3;
            const bf16* g = xT + (size_t)(b * T_IN + t0 + r) * FILT + ks * KS + seg * 8;
            ASYNC_CP16(g, &xs_s[buf][r * XPITCH + seg * 8]);
        }
    };
    auto issue_pv = [&](int ks, int buf) {
        if (tid < (NT + 2) * 4) {
            int r = tid >> 2, seg = tid & 3;
            const bf16* g = prevT + (size_t)(b * (T_IN + 2) + t0 + r) * C_INCH + ks * KS + seg * 8;
            ASYNC_CP16(g, &pv_s[buf][r * XPITCH + seg * 8]);
        }
    };

    // ---- loop 1: out += (W_mod * s) x, K = 512 ----
    issue_x(0, 0);
    for (int ks = 0; ks < 16; ++ks) {
        if (ks + 1 < 16) { issue_x(ks + 1, (ks + 1) & 1); ASYNC_WAIT(1); }
        else             { ASYNC_WAIT(0); }
        __syncthreads();
        union { uint4 q[2]; v16bf v; } fb, fa0, fa1;
        const uint4* bp = reinterpret_cast<const uint4*>(&xs_s[ks & 1][brow * XPITCH + bcol]);
        fb.q[0] = bp[0]; fb.q[1] = bp[2];
        const uint4* ap0 = reinterpret_cast<const uint4*>(
            wmodFb + ((size_t)(((b * 32 + ot0 + 0) * 16 + ks) * 32 + lane) << 4));
        fa0.q[0] = ap0[0]; fa0.q[1] = ap0[1];
        const uint4* ap1 = reinterpret_cast<const uint4*>(
            wmodFb + ((size_t)(((b * 32 + ot0 + 1) * 16 + ks) * 32 + lane) << 4));
        fa1.q[0] = ap1[0]; fa1.q[1] = ap1[1];
        acc0 = __builtin_amdgcn_wmma_f32_16x16x32_bf16(false, fa0.v, false, fb.v, (short)0, acc0, false, false);
        acc1 = __builtin_amdgcn_wmma_f32_16x16x32_bf16(false, fa1.v, false, fb.v, (short)0, acc1, false, false);
        __syncthreads();
    }

    // ---- loop 2: out += conv1d(prev_rgb), K = 256, taps {-1,0,+1} ----
    issue_pv(0, 0);
    for (int ks = 0; ks < 8; ++ks) {
        if (ks + 1 < 8) { issue_pv(ks + 1, (ks + 1) & 1); ASYNC_WAIT(1); }
        else            { ASYNC_WAIT(0); }
        __syncthreads();
        union { uint4 q[2]; v16bf v; } fb, fa0, fa1;
        #pragma unroll
        for (int tap = 0; tap < 3; ++tap) {
            const uint4* bp = reinterpret_cast<const uint4*>(
                &pv_s[ks & 1][(brow + tap) * XPITCH + bcol]);
            fb.q[0] = bp[0]; fb.q[1] = bp[2];
            const uint4* ap0 = reinterpret_cast<const uint4*>(
                wconvF + ((size_t)(((tap * 32 + ot0 + 0) * 8 + ks) * 32 + lane) << 4));
            fa0.q[0] = ap0[0]; fa0.q[1] = ap0[1];
            const uint4* ap1 = reinterpret_cast<const uint4*>(
                wconvF + ((size_t)(((tap * 32 + ot0 + 1) * 8 + ks) * 32 + lane) << 4));
            fa1.q[0] = ap1[0]; fa1.q[1] = ap1[1];
            acc0 = __builtin_amdgcn_wmma_f32_16x16x32_bf16(false, fa0.v, false, fb.v, (short)0, acc0, false, false);
            acc1 = __builtin_amdgcn_wmma_f32_16x16x32_bf16(false, fa1.v, false, fb.v, (short)0, acc1, false, false);
        }
        __syncthreads();
    }

    // ---- spill accumulators to LDS (D layout: n=lane%16, m=r+8*(lane/16)) ----
    {
        const int m_base = o_sub * 32;
        const int trow = tn + l16;
        #pragma unroll
        for (int r = 0; r < 8; ++r) {
            int m = r + (lh << 3);
            mid_s[trow * MPITCH + m_base + m]      = acc0[r];
            mid_s[trow * MPITCH + m_base + 16 + m] = acc1[r];
        }
    }
    __syncthreads();

    // ---- fused 2x linear upsample (interior columns) + edge export ----
    const int j_l = tid & 63;
    const int o_start = tid >> 6;          // 4 o-rows per pass
    for (int o_loc = o_start; o_loc < MT; o_loc += 4) {
        const int o_g = ob * MT + o_loc;
        const float bv = bias[o_g];
        if (j_l == 0) {
            edgeF[(size_t)(b * NTB + tb) * FILT + o_g] = mid_s[0 * MPITCH + o_loc] + bv;
        } else if (j_l == 63) {
            edgeL[(size_t)(b * NTB + tb) * FILT + o_g] = mid_s[31 * MPITCH + o_loc] + bv;
        } else {
            const int i0 = (j_l - 1) >> 1;
            const float m0 = mid_s[i0 * MPITCH + o_loc] + bv;
            const float m1 = mid_s[(i0 + 1) * MPITCH + o_loc] + bv;
            const float v = (j_l & 1) ? (0.75f * m0 + 0.25f * m1)
                                      : (0.25f * m0 + 0.75f * m1);
            out[(size_t)(b * FILT + o_g) * T_OUT + (size_t)tb * 64 + j_l] = v;
        }
    }
}

// ---------------------------------------------------------------------------
// Cross-tile boundary columns of the upsample (2 per tile).
// ---------------------------------------------------------------------------
__global__ void k_cleanup(const float* __restrict__ edgeF,
                          const float* __restrict__ edgeL,
                          float* __restrict__ out) {
    int idx = blockIdx.x * blockDim.x + threadIdx.x;
    if (idx >= B_SZ * FILT * NTB) return;
    int tb = idx & 255;
    int o  = (idx >> 8) & 511;
    int b  = idx >> 17;
    float eF = edgeF[(size_t)(b * NTB + tb) * FILT + o];
    float eL = edgeL[(size_t)(b * NTB + tb) * FILT + o];
    float v0 = (tb == 0) ? eF
                         : 0.25f * edgeL[(size_t)(b * NTB + tb - 1) * FILT + o] + 0.75f * eF;
    float v1 = (tb == NTB - 1) ? eL
                               : 0.75f * eL + 0.25f * edgeF[(size_t)(b * NTB + tb + 1) * FILT + o];
    size_t base = (size_t)(b * FILT + o) * T_OUT + (size_t)tb * 64;
    out[base + 0]  = v0;
    out[base + 63] = v1;
}

// ---------------------------------------------------------------------------
extern "C" void kernel_launch(void* const* d_in, const int* in_sizes, int n_in,
                              void* d_out, int out_size, void* d_ws, size_t ws_size,
                              hipStream_t stream) {
    const float* x      = (const float*)d_in[0];
    const float* prevr  = (const float*)d_in[1];
    const float* istyle = (const float*)d_in[2];
    const float* Wsty   = (const float*)d_in[3];
    const float* bsty   = (const float*)d_in[4];
    const float* Wmod   = (const float*)d_in[5];
    const float* Wconv  = (const float*)d_in[6];
    const float* bconv  = (const float*)d_in[7];
    float* out = (float*)d_out;

    char* ws = (char*)d_ws;
    size_t off = 0;
    auto walloc = [&](size_t bytes) -> void* {
        void* p = ws + off;
        off = (off + bytes + 255) & ~(size_t)255;
        return p;
    };
    float* s      = (float*)walloc((size_t)B_SZ * FILT * sizeof(float));
    bf16*  wmodFb = (bf16*)walloc((size_t)B_SZ * 32 * 16 * 32 * 16 * sizeof(bf16));
    bf16*  wconvF = (bf16*)walloc((size_t)3 * 32 * 8 * 32 * 16 * sizeof(bf16));
    bf16*  xT     = (bf16*)walloc((size_t)B_SZ * T_IN * FILT * sizeof(bf16));
    bf16*  prevT  = (bf16*)walloc((size_t)B_SZ * (T_IN + 2) * C_INCH * sizeof(bf16));
    float* edgeF  = (float*)walloc((size_t)B_SZ * NTB * FILT * sizeof(float));
    float* edgeL  = (float*)walloc((size_t)B_SZ * NTB * FILT * sizeof(float));
    (void)ws_size; (void)in_sizes; (void)n_in; (void)out_size;

    k_style<<<(B_SZ * FILT + 255) / 256, 256, 0, stream>>>(istyle, Wsty, bsty, s);
    k_pack_wmod<<<(B_SZ * 32 * 16 * 32 * 16) / 256, 256, 0, stream>>>(Wmod, s, wmodFb);
    k_pack_wconv<<<(3 * 32 * 8 * 32 * 16) / 256, 256, 0, stream>>>(Wconv, wconvF);

    dim3 gtx(T_IN / 32, FILT / 32, B_SZ);
    k_tr_x<<<gtx, 256, 0, stream>>>(x, xT);
    dim3 gtp(T_IN / 32, C_INCH / 32, B_SZ);
    k_tr_prev<<<gtp, 256, 0, stream>>>(prevr, prevT);
    k_zero_pad<<<(B_SZ * 2 * C_INCH + 255) / 256, 256, 0, stream>>>(prevT);

    dim3 grid(NTB, FILT / MT, B_SZ);   // (256, 4, 8)
    k_main<<<grid, 256, 0, stream>>>(xT, prevT, wmodFb, wconvF, bconv, out, edgeF, edgeL);

    k_cleanup<<<(B_SZ * FILT * NTB) / 256, 256, 0, stream>>>(edgeF, edgeL, out);
}
